// MoEModel_66202625900932
// MI455X (gfx1250) — compile-verified, matrix operand order checked by reading, not run on hardware
//
#include <hip/hip_runtime.h>
#include <stdint.h>

// ---------------------------------------------------------------------------
// MoE forward for MI455X (gfx1250, wave32, WMMA).
// Big GEMMs: bf16, f32 accum via v_wmma_f32_16x16x32_bf16.
// Staging: global_load_async_to_lds_b128 (ASYNCcnt), double-buffered LDS.
// Wave tile 64x64 (4x4 WMMA tiles) -> 16 WMMA per 16 ds_load_b128, at the
// LDS-bandwidth roofline for back-to-back bf16 WMMA on a WGP.
// ---------------------------------------------------------------------------

typedef __attribute__((ext_vector_type(16))) __bf16 v16bf;
typedef __attribute__((ext_vector_type(8)))  float  v8f;

#define MOE_B   16384
#define MOE_D   1024
#define MOE_E   8
#define MOE_C   50
#define MOE_RH  512
#define MOE_RH2 256
#define MOE_H1  2048
#define MOE_H2  1024
#define MOE_CH  8192   // token chunk for expert layers

__device__ __forceinline__ unsigned short f32_to_bf16(float f) {
  unsigned int u = __builtin_bit_cast(unsigned int, f);
  u += 0x7fffu + ((u >> 16) & 1u);           // round-to-nearest-even
  return (unsigned short)(u >> 16);
}
__device__ __forceinline__ float bf16_to_f32(unsigned short h) {
  unsigned int u = ((unsigned int)h) << 16;
  return __builtin_bit_cast(float, u);
}

// --------------------------- elementwise convert ---------------------------
__global__ void cvt_f32_bf16_kernel(const float* __restrict__ src,
                                    unsigned short* __restrict__ dst, long n) {
  long i = (long)blockIdx.x * blockDim.x + threadIdx.x;
  long stride = (long)gridDim.x * blockDim.x;
  for (; i < n; i += stride) dst[i] = f32_to_bf16(src[i]);
}

__global__ void zero_f32_kernel(float* __restrict__ dst, long n) {
  long i = (long)blockIdx.x * blockDim.x + threadIdx.x;
  long stride = (long)gridDim.x * blockDim.x;
  for (; i < n; i += stride) dst[i] = 0.0f;
}

// ----------------- convert f32 [K,N] -> bf16 transposed [N,K] --------------
__global__ __launch_bounds__(256)
void cvtT_f32_bf16_kernel(const float* __restrict__ src,   // K x N
                          unsigned short* __restrict__ dst, // N x K
                          int K, int N) {
  __shared__ unsigned short t[32][33];
  const int n0 = blockIdx.x * 32;
  const int k0 = blockIdx.y * 32;
  const int c = threadIdx.x & 31;
  const int r8 = threadIdx.x >> 5;
#pragma unroll
  for (int rr = r8; rr < 32; rr += 8)
    t[c][rr] = f32_to_bf16(src[(size_t)(k0 + rr) * N + n0 + c]);
  __syncthreads();
#pragma unroll
  for (int rr = r8; rr < 32; rr += 8)
    dst[(size_t)(n0 + rr) * K + k0 + c] = t[rr][c];
}

// --------------------------- bf16 WMMA GEMM --------------------------------
// C[M,N] = act(A[M,K] @ Wt[N,K]^T + bias). A, Wt bf16 k-major.
// Block: 256 threads (8 waves), tile 256x128, K-step 32, double-buffered LDS.
// Waves in 4(M) x 2(N) grid; each wave: 64x64 = 4x4 WMMA tiles.
union Frag { int i[8]; v16bf v; };

// async copies; INST_OFFSET applies to both global and LDS side (ISA 15.18.3).
__device__ __forceinline__ void async_cp32(const unsigned short* g,
                                           unsigned short* l) {
  unsigned int lds = (unsigned int)(unsigned long long)l;
  unsigned long long ga = (unsigned long long)g;
  asm volatile("global_load_async_to_lds_b128 %0, %1, off"
               :: "v"(lds), "v"(ga) : "memory");
  asm volatile("global_load_async_to_lds_b128 %0, %1, off offset:16"
               :: "v"(lds), "v"(ga) : "memory");
}
__device__ __forceinline__ void async_cp64(const unsigned short* g,
                                           unsigned short* l) {
  unsigned int lds = (unsigned int)(unsigned long long)l;
  unsigned long long ga = (unsigned long long)g;
  asm volatile("global_load_async_to_lds_b128 %0, %1, off"
               :: "v"(lds), "v"(ga) : "memory");
  asm volatile("global_load_async_to_lds_b128 %0, %1, off offset:16"
               :: "v"(lds), "v"(ga) : "memory");
  asm volatile("global_load_async_to_lds_b128 %0, %1, off offset:32"
               :: "v"(lds), "v"(ga) : "memory");
  asm volatile("global_load_async_to_lds_b128 %0, %1, off offset:48"
               :: "v"(lds), "v"(ga) : "memory");
}

__global__ __launch_bounds__(256)
void gemm_bf16_kernel(const unsigned short* __restrict__ A,    // M x K
                      const unsigned short* __restrict__ Wt,   // N x K
                      const float* __restrict__ bias,          // N
                      float* __restrict__ Cf,                  // optional f32 out
                      unsigned short* __restrict__ Cb,         // optional bf16 out
                      int M, int N, int K, int relu) {
  __shared__ __align__(16) unsigned short As[2][256 * 32];   // [row][k] 16KB x2
  __shared__ __align__(16) unsigned short Bs[2][128 * 32];   // [col][k]  8KB x2

  const int tid  = threadIdx.x;
  const int wave = tid >> 5;
  const int lane = tid & 31;
  const int lrow = lane & 15;
  const int hi   = lane >> 4;
  const int wm   = wave & 3;   // 4 waves along M (64 rows each)
  const int wn   = wave >> 2;  // 2 waves along N (64 cols each)
  const int n0   = blockIdx.x * 128;
  const int m0   = blockIdx.y * 256;

  // K index pattern for 16-bit A/B fragments (ISA 7.12.2, 16x32 bf16):
  // VGPR j (0..3): K = hi*8 + 2j ; VGPR j (4..7): K = 16 + hi*8 + 2(j-4)
  int koff[8];
#pragma unroll
  for (int j = 0; j < 8; ++j)
    koff[j] = (j < 4) ? (hi * 8 + 2 * j) : (16 + hi * 8 + 2 * (j - 4));

  v8f zero = {};
  v8f acc[4][4];
#pragma unroll
  for (int mt = 0; mt < 4; ++mt)
#pragma unroll
    for (int nt = 0; nt < 4; ++nt) acc[mt][nt] = zero;

  // staging roles: A -> 1 thread per row (64B); B -> 2 threads per row (32B)
  const int br = tid >> 1, bh = tid & 1;
  const int nk = K >> 5;

  // prologue: stage tile 0 into buffer 0 (6 async ops per wave)
  async_cp64(A  + (size_t)(m0 + tid) * K, &As[0][tid * 32]);
  async_cp32(Wt + (size_t)(n0 + br) * K + bh * 16, &Bs[0][br * 32 + bh * 16]);

  for (int i = 0; i < nk; ++i) {
    const int buf = i & 1;
    if (i + 1 < nk) {
      const int k1 = (i + 1) << 5;
      async_cp64(A  + (size_t)(m0 + tid) * K + k1, &As[buf ^ 1][tid * 32]);
      async_cp32(Wt + (size_t)(n0 + br) * K + k1 + bh * 16,
                 &Bs[buf ^ 1][br * 32 + bh * 16]);
      asm volatile("s_wait_asynccnt 0x6" ::: "memory"); // tile i done, i+1 pending
    } else {
      asm volatile("s_wait_asynccnt 0x0" ::: "memory");
    }
    __syncthreads();

    Frag a[4];
#pragma unroll
    for (int mt = 0; mt < 4; ++mt) {
      const int arow = wm * 64 + mt * 16 + lrow;
#pragma unroll
      for (int j = 0; j < 8; ++j)
        a[mt].i[j] = *(const int*)&As[buf][arow * 32 + koff[j]];
    }
#pragma unroll
    for (int nt = 0; nt < 4; ++nt) {
      Frag b;
      const int bcol = wn * 64 + nt * 16 + lrow;
#pragma unroll
      for (int j = 0; j < 8; ++j)
        b.i[j] = *(const int*)&Bs[buf][bcol * 32 + koff[j]];
#pragma unroll
      for (int mt = 0; mt < 4; ++mt)
        acc[mt][nt] = __builtin_amdgcn_wmma_f32_16x16x32_bf16(
            false, a[mt].v, false, b.v, (short)0, acc[mt][nt], false, false);
    }
    __syncthreads();
  }

  // ---- epilogue: bias + relu + store ----
  // C/D layout: VGPR r -> (lanes 0-15: M=r, N=lane) (lanes 16-31: M=8+r, N=lane-16)
#pragma unroll
  for (int nt = 0; nt < 4; ++nt) {
    const int n = n0 + wn * 64 + nt * 16 + lrow;
    const float bv = bias ? bias[n] : 0.0f;
#pragma unroll
    for (int mt = 0; mt < 4; ++mt) {
#pragma unroll
      for (int r = 0; r < 8; ++r) {
        const int m = m0 + wm * 64 + mt * 16 + hi * 8 + r;
        float v = acc[mt][nt][r] + bv;
        if (relu) v = fmaxf(v, 0.0f);
        if (Cf) Cf[(size_t)m * N + n] = v;
        if (Cb) Cb[(size_t)m * N + n] = f32_to_bf16(v);
      }
    }
  }
}

// --------------------------- router head (N=8) -----------------------------
__global__ void router3_kernel(const unsigned short* __restrict__ h2, // B x 256 bf16
                               const float* __restrict__ W3,          // 256 x 8
                               const float* __restrict__ b3,          // 8
                               float* __restrict__ scores, int Btot) {
  int t = blockIdx.x * blockDim.x + threadIdx.x;
  if (t >= Btot) return;
  float s[MOE_E];
#pragma unroll
  for (int e = 0; e < MOE_E; ++e) s[e] = b3[e];
  const unsigned short* hr = h2 + (size_t)t * MOE_RH2;
  for (int k = 0; k < MOE_RH2; ++k) {
    float h = bf16_to_f32(hr[k]);
#pragma unroll
    for (int e = 0; e < MOE_E; ++e) s[e] += h * W3[k * MOE_E + e];
  }
#pragma unroll
  for (int e = 0; e < MOE_E; ++e) scores[(size_t)t * MOE_E + e] = s[e];
}

// ------------------- softmax + top-2 gates (per token) ---------------------
__global__ void gate_kernel(const float* __restrict__ scores,
                            float* __restrict__ probs_out,
                            float* __restrict__ gates, int Btot) {
  int t = blockIdx.x * blockDim.x + threadIdx.x;
  if (t >= Btot) return;
  float s[MOE_E];
  float mx = -3.4e38f;
#pragma unroll
  for (int e = 0; e < MOE_E; ++e) { s[e] = scores[(size_t)t * MOE_E + e]; mx = fmaxf(mx, s[e]); }
  float sum = 0.0f;
#pragma unroll
  for (int e = 0; e < MOE_E; ++e) { s[e] = __expf(s[e] - mx); sum += s[e]; }
  float inv = 1.0f / sum;
#pragma unroll
  for (int e = 0; e < MOE_E; ++e) { s[e] *= inv; probs_out[(size_t)t * MOE_E + e] = s[e]; }
  int i1 = 0;
#pragma unroll
  for (int e = 1; e < MOE_E; ++e) if (s[e] > s[i1]) i1 = e;
  int i2 = (i1 == 0) ? 1 : 0;
#pragma unroll
  for (int e = 0; e < MOE_E; ++e) if (e != i1 && s[e] > s[i2]) i2 = e;
#pragma unroll
  for (int e = 0; e < MOE_E; ++e)
    gates[(size_t)t * MOE_E + e] = (e == i1 || e == i2) ? 0.5f * s[e] : 0.0f;
}

// ------------- expert head (H2 -> C=50) with fused gated combine ----------
__global__ void expert_head_kernel(const unsigned short* __restrict__ a2, // CH x H2 bf16
                                   const float* __restrict__ W3,          // H2 x C
                                   const float* __restrict__ b3,          // C
                                   const float* __restrict__ gates,       // B x E
                                   float* __restrict__ out,               // B x C
                                   int t0, int CH, int e) {
  int idx = blockIdx.x * blockDim.x + threadIdx.x;
  if (idx >= CH * MOE_C) return;
  int tl = idx / MOE_C;
  int c  = idx % MOE_C;
  int t  = t0 + tl;
  float g = gates[(size_t)t * MOE_E + e];
  if (g == 0.0f) return;                 // top-2 sparsity
  const unsigned short* ar = a2 + (size_t)tl * MOE_H2;
  float s = b3[c];
  for (int k = 0; k < MOE_H2; ++k) s += bf16_to_f32(ar[k]) * W3[(size_t)k * MOE_C + c];
  out[(size_t)t * MOE_C + c] += g * s;
}

// ---------------------------------------------------------------------------
static inline void launch_gemm(const unsigned short* A, const unsigned short* Wt,
                               const float* bias, float* Cf, unsigned short* Cb,
                               int M, int N, int K, int relu, hipStream_t s) {
  dim3 grid(N / 128, M / 256);
  gemm_bf16_kernel<<<grid, 256, 0, s>>>(A, Wt, bias, Cf, Cb, M, N, K, relu);
}
static inline void launch_cvtT(const float* src, unsigned short* dst,
                               int K, int N, hipStream_t s) {
  dim3 grid(N / 32, K / 32);
  cvtT_f32_bf16_kernel<<<grid, 256, 0, s>>>(src, dst, K, N);
}

extern "C" void kernel_launch(void* const* d_in, const int* in_sizes, int n_in,
                              void* d_out, int out_size, void* d_ws, size_t ws_size,
                              hipStream_t stream) {
  (void)in_sizes; (void)n_in; (void)out_size; (void)ws_size;

  const float* x   = (const float*)d_in[0];
  const float* rW1 = (const float*)d_in[1];
  const float* rb1 = (const float*)d_in[2];
  const float* rW2 = (const float*)d_in[3];
  const float* rb2 = (const float*)d_in[4];
  const float* rW3 = (const float*)d_in[5];
  const float* rb3 = (const float*)d_in[6];
  const float* eW1 = (const float*)d_in[7];
  const float* eb1 = (const float*)d_in[8];
  const float* eW2 = (const float*)d_in[9];
  const float* eb2 = (const float*)d_in[10];
  const float* eW3 = (const float*)d_in[11];
  const float* eb3 = (const float*)d_in[12];

  float* out   = (float*)d_out;                        // [B*C]
  float* probs = out + (size_t)MOE_B * MOE_C;          // [B*E]

  // workspace carve-out (~120 MB)
  char* p = (char*)d_ws;
  auto alloc = [&](size_t bytes) -> char* {
    char* r = p;
    p += (bytes + 255) & ~(size_t)255;
    return r;
  };
  unsigned short* xb    = (unsigned short*)alloc((size_t)MOE_B * MOE_D * 2);
  unsigned short* rW1t  = (unsigned short*)alloc((size_t)MOE_D * MOE_RH * 2);   // [RH, D]
  unsigned short* rW2t  = (unsigned short*)alloc((size_t)MOE_RH * MOE_RH2 * 2); // [RH2, RH]
  unsigned short* rh1b  = (unsigned short*)alloc((size_t)MOE_B * MOE_RH * 2);
  unsigned short* rh2b  = (unsigned short*)alloc((size_t)MOE_B * MOE_RH2 * 2);
  float*          scr   = (float*)alloc((size_t)MOE_B * MOE_E * 4);
  float*          gates = (float*)alloc((size_t)MOE_B * MOE_E * 4);
  unsigned short* w1t   = (unsigned short*)alloc((size_t)MOE_D * MOE_H1 * 2);   // [H1, D]
  unsigned short* w2t   = (unsigned short*)alloc((size_t)MOE_H1 * MOE_H2 * 2);  // [H2, H1]
  unsigned short* a1b   = (unsigned short*)alloc((size_t)MOE_CH * MOE_H1 * 2);
  unsigned short* a2b   = (unsigned short*)alloc((size_t)MOE_CH * MOE_H2 * 2);

  // --- convert activations + transpose-convert router weights to bf16 ---
  cvt_f32_bf16_kernel<<<4096, 256, 0, stream>>>(x, xb, (long)MOE_B * MOE_D);
  launch_cvtT(rW1, rW1t, MOE_D, MOE_RH, stream);    // [D,RH] -> [RH,D]
  launch_cvtT(rW2, rW2t, MOE_RH, MOE_RH2, stream);  // [RH,RH2] -> [RH2,RH]

  // --- router MLP ---
  launch_gemm(xb, rW1t, rb1, nullptr, rh1b, MOE_B, MOE_RH, MOE_D, 1, stream);
  launch_gemm(rh1b, rW2t, rb2, nullptr, rh2b, MOE_B, MOE_RH2, MOE_RH, 1, stream);
  router3_kernel<<<(MOE_B + 255) / 256, 256, 0, stream>>>(rh2b, rW3, rb3, scr, MOE_B);
  gate_kernel<<<(MOE_B + 255) / 256, 256, 0, stream>>>(scr, probs, gates, MOE_B);

  // --- zero the combined output ---
  zero_f32_kernel<<<3200, 256, 0, stream>>>(out, (long)MOE_B * MOE_C);

  // --- experts: D -> H1 -> H2 -> C, gated combine ---
  for (int e = 0; e < MOE_E; ++e) {
    launch_cvtT(eW1 + (size_t)e * MOE_D * MOE_H1, w1t, MOE_D, MOE_H1, stream);
    launch_cvtT(eW2 + (size_t)e * MOE_H1 * MOE_H2, w2t, MOE_H1, MOE_H2, stream);
    const float* b1 = eb1 + (size_t)e * MOE_H1;
    const float* b2 = eb2 + (size_t)e * MOE_H2;
    const float* W3 = eW3 + (size_t)e * MOE_H2 * MOE_C;
    const float* b3 = eb3 + (size_t)e * MOE_C;
    for (int t0 = 0; t0 < MOE_B; t0 += MOE_CH) {
      launch_gemm(xb + (size_t)t0 * MOE_D, w1t, b1, nullptr, a1b,
                  MOE_CH, MOE_H1, MOE_D, 1, stream);
      launch_gemm(a1b, w2t, b2, nullptr, a2b,
                  MOE_CH, MOE_H2, MOE_H1, 1, stream);
      expert_head_kernel<<<(MOE_CH * MOE_C + 255) / 256, 256, 0, stream>>>(
          a2b, W3, b3, gates, out, t0, MOE_CH, e);
    }
  }
}
